// WfCausalConv3dAttnBlock_7567732376027
// MI455X (gfx1250) — matrix operand order repaired and checked by reading, not compiled, not run on hardware
//
#include <hip/hip_runtime.h>

// Problem constants (from reference)
#define CC 512        // channels
#define SS 16384      // t*h*w per batch
#define HW 1024       // h*w (attention sequence length)
#define NBATCH 2
#define NBT 32        // b*t attention batches

// LDS tile strides.
// LDA_K: 34 halves = 17-dword row stride -> conflict-free A fragment b32 reads.
// LDB_N: 136 halves = 272B row stride (multiple of 16B -> legal target for
//        global_load_async_to_lds_b128 scatter into padded rows).
#define LDA_K 34
#define LDB_N 136

typedef __attribute__((ext_vector_type(16))) _Float16 v16h;
typedef __attribute__((ext_vector_type(8)))  float    v8f;

// ---------------------------------------------------------------------------
// CDNA5 async memory->LDS copy (16B per lane), tracked by ASYNCcnt.
// lds_off: low 32 bits of the flat address of a __shared__ location (= LDS
// byte offset per the aperture rules). gaddr: 64-bit global address.
// ---------------------------------------------------------------------------
__device__ __forceinline__ void async_copy_b128(unsigned int lds_off,
                                                unsigned long long gaddr) {
  asm volatile("global_load_async_to_lds_b128 %0, %1, off"
               :: "v"(lds_off), "v"(gaddr)
               : "memory");
}
__device__ __forceinline__ void async_wait0() {
  asm volatile("s_wait_asynccnt 0" ::: "memory");
}
__device__ __forceinline__ unsigned int lds_addr(const void* p) {
  return (unsigned int)(size_t)p;
}

// ---------------------------------------------------------------------------
// Wave-level 32x64 tile: gather A(16x32)/B(32x16) fragments from LDS per the
// CDNA5 WMMA VGPR layouts and issue 8 v_wmma_f32_16x16x32_f16 per K-step.
// ---------------------------------------------------------------------------
__device__ __forceinline__ void wmma_tile_compute(const _Float16* __restrict__ As,
                                                  const _Float16* __restrict__ Bs,
                                                  v8f acc[2][4], int wm, int wn,
                                                  int lane) {
  const int half = lane >> 4;
  const int mr = lane & 15;
  v16h a[2];
  v16h b[4];
#pragma unroll
  for (int mi = 0; mi < 2; ++mi) {
    const _Float16* Arow = As + (wm + mi * 16 + mr) * LDA_K;
#pragma unroll
    for (int v = 0; v < 8; ++v) {
      // A 16x32: lanes0-15 v<4: K=2v,2v+1 ; v>=4: K=2v+8,2v+9 ; +8 for lanes16-31
      const int kl = 2 * v + ((v >= 4) ? 8 : 0) + 8 * half;
      a[mi][2 * v]     = Arow[kl];
      a[mi][2 * v + 1] = Arow[kl + 1];
    }
  }
#pragma unroll
  for (int ni = 0; ni < 4; ++ni) {
    const int n = wn + ni * 16 + mr;
#pragma unroll
    for (int v = 0; v < 8; ++v) {
      // B 32x16: lanes0-15 v<4: K=2v,2v+1 ; v>=4: K=2v+8,2v+9 ; +8 for lanes16-31
      const int kl = 2 * v + ((v >= 4) ? 8 : 0) + 8 * half;
      b[ni][2 * v]     = Bs[kl * LDB_N + n];
      b[ni][2 * v + 1] = Bs[(kl + 1) * LDB_N + n];
    }
  }
#pragma unroll
  for (int mi = 0; mi < 2; ++mi)
#pragma unroll
    for (int ni = 0; ni < 4; ++ni)
      acc[mi][ni] = __builtin_amdgcn_wmma_f32_16x16x32_f16(
          false, a[mi], false, b[ni], (short)0, acc[mi][ni], false, false);
}

// ---------------------------------------------------------------------------
// GroupNorm statistics: one block per (batch, group); each group's 16 channels
// are a contiguous 16*SS-float span in [b][c][s] layout.
// ---------------------------------------------------------------------------
__global__ __launch_bounds__(256) void gn_stats_kernel(const float* __restrict__ x,
                                                       float* __restrict__ stats) {
  const int bg = blockIdx.x;  // 0..63
  const int tid = threadIdx.x;
  const float4* xp = reinterpret_cast<const float4*>(x + (size_t)bg * 16 * SS);
  const int n4 = (16 * SS) / 4;
  float s1 = 0.f, s2 = 0.f;
  for (int i = tid; i < n4; i += 256) {
    float4 v = xp[i];
    s1 += v.x + v.y + v.z + v.w;
    s2 += v.x * v.x + v.y * v.y + v.z * v.z + v.w * v.w;
  }
  __shared__ float r1[256], r2[256];
  r1[tid] = s1; r2[tid] = s2;
  __syncthreads();
  for (int off = 128; off > 0; off >>= 1) {
    if (tid < off) { r1[tid] += r1[tid + off]; r2[tid] += r2[tid + off]; }
    __syncthreads();
  }
  if (tid == 0) {
    const float inv_n = 1.0f / (16.0f * SS);
    const float mean = r1[0] * inv_n;
    const float var = r2[0] * inv_n - mean * mean;
    stats[2 * bg] = mean;
    stats[2 * bg + 1] = rsqrtf(var + 1e-6f);
  }
}

// Apply GroupNorm and emit f16 hn in [b][c][s] layout.
__global__ __launch_bounds__(256) void gn_apply_kernel(
    const float* __restrict__ x, const float* __restrict__ gamma,
    const float* __restrict__ beta, const float* __restrict__ stats,
    _Float16* __restrict__ hn) {
  const size_t i4 = ((size_t)blockIdx.x * 256 + threadIdx.x) * 4;
  const int c = (int)((i4 / SS) % CC);
  const int bg = (int)(i4 / ((size_t)16 * SS));  // = b*32 + group
  const float mean = stats[2 * bg];
  const float rstd = stats[2 * bg + 1];
  const float g = gamma[c] * rstd;
  const float bt = beta[c] - mean * g;
  const float4 v = *reinterpret_cast<const float4*>(x + i4);
  hn[i4 + 0] = (_Float16)(v.x * g + bt);
  hn[i4 + 1] = (_Float16)(v.y * g + bt);
  hn[i4 + 2] = (_Float16)(v.z * g + bt);
  hn[i4 + 3] = (_Float16)(v.w * g + bt);
}

// ---------------------------------------------------------------------------
// Pointwise conv GEMM: Out[o][s] = sum_c W[o][c] * B[c][s] + bias[o]
// OUT16=true  -> f16 store (q/k/v paths)
// OUT16=false -> fp32 store with residual add (projection path)
// B tile staged via GLOBAL_LOAD_ASYNC_TO_LDS_B128 (ASYNCcnt).
// ---------------------------------------------------------------------------
template <bool OUT16>
__global__ __launch_bounds__(256) void gemm_conv_kernel(
    const float* __restrict__ W, const float* __restrict__ bias,
    const _Float16* __restrict__ Bmat, _Float16* __restrict__ out16,
    float* __restrict__ out32, const float* __restrict__ resid) {
  __shared__ __align__(16) _Float16 As[128 * LDA_K];
  __shared__ __align__(16) _Float16 Bs[32 * LDB_N];
  const int tid = threadIdx.x;
  const int lane = tid & 31;
  const int wid = tid >> 5;
  const int wm = (wid >> 1) * 32;
  const int wn = (wid & 1) * 64;
  const int n0 = blockIdx.x * 128;  // spatial
  const int m0 = blockIdx.y * 128;  // out channel
  const int z = blockIdx.z;         // batch
  const _Float16* Bp = Bmat + (size_t)z * CC * SS;

  v8f acc[2][4];
#pragma unroll
  for (int mi = 0; mi < 2; ++mi)
#pragma unroll
    for (int ni = 0; ni < 4; ++ni)
#pragma unroll
      for (int j = 0; j < 8; ++j) acc[mi][ni][j] = 0.0f;

  const int amm = tid >> 1;           // 0..127 (M rows of A tile)
  const int akk0 = (tid & 1) * 16;    // K half
  const int bkk = tid >> 3;           // 0..31 (K rows of B tile)
  const int bnn0 = (tid & 7) * 16;    // N chunk

  for (int c0 = 0; c0 < CC; c0 += 32) {
    __syncthreads();
    {  // stage B: hn[c][s], contiguous N; direct async copy into padded LDS rows
      const _Float16* src = Bp + (size_t)(c0 + bkk) * SS + n0 + bnn0;
      const unsigned int l0 = lds_addr(Bs + bkk * LDB_N + bnn0);
      async_copy_b128(l0, (unsigned long long)(size_t)src);
      async_copy_b128(l0 + 16, (unsigned long long)(size_t)(src + 8));
    }
    {  // stage A: W[o][c] fp32 -> f16, row-major [mm][kk] (needs conversion)
      const float4* w4 =
          reinterpret_cast<const float4*>(W + (size_t)(m0 + amm) * CC + c0 + akk0);
      _Float16* dst = As + amm * LDA_K + akk0;
#pragma unroll
      for (int j = 0; j < 4; ++j) {
        const float4 f = w4[j];
        dst[4 * j + 0] = (_Float16)f.x;
        dst[4 * j + 1] = (_Float16)f.y;
        dst[4 * j + 2] = (_Float16)f.z;
        dst[4 * j + 3] = (_Float16)f.w;
      }
    }
    async_wait0();
    __syncthreads();
    wmma_tile_compute(As, Bs, acc, wm, wn, lane);
  }

  const int half = lane >> 4;
  const int nr = lane & 15;
#pragma unroll
  for (int mi = 0; mi < 2; ++mi)
#pragma unroll
    for (int ni = 0; ni < 4; ++ni)
#pragma unroll
      for (int r = 0; r < 8; ++r) {
        const int row = m0 + wm + mi * 16 + r + 8 * half;  // o
        const int col = n0 + wn + ni * 16 + nr;            // s
        const size_t idx = (size_t)z * CC * SS + (size_t)row * SS + col;
        const float val = acc[mi][ni][r] + bias[row];
        if (OUT16)
          out16[idx] = (_Float16)val;
        else
          out32[idx] = resid[idx] + val;
      }
}

// ---------------------------------------------------------------------------
// S[qi][ki] = scale * sum_c Q[c][qi] * K[c][ki]   (per bt batch)
// B tile staged via async-to-LDS; A tile is a transpose (register path).
// ---------------------------------------------------------------------------
__global__ __launch_bounds__(256) void gemm_qk_kernel(
    const _Float16* __restrict__ qb, const _Float16* __restrict__ kb,
    float* __restrict__ Sb) {
  __shared__ __align__(16) _Float16 As[128 * LDA_K];
  __shared__ __align__(16) _Float16 Bs[32 * LDB_N];
  const int tid = threadIdx.x;
  const int lane = tid & 31;
  const int wid = tid >> 5;
  const int wm = (wid >> 1) * 32;
  const int wn = (wid & 1) * 64;
  const int n0 = blockIdx.x * 128;  // key index
  const int m0 = blockIdx.y * 128;  // query index
  const int zz = blockIdx.z;        // bt
  const size_t base = (size_t)(zz >> 4) * CC * SS + (size_t)(zz & 15) * HW;

  v8f acc[2][4];
#pragma unroll
  for (int mi = 0; mi < 2; ++mi)
#pragma unroll
    for (int ni = 0; ni < 4; ++ni)
#pragma unroll
      for (int j = 0; j < 8; ++j) acc[mi][ni][j] = 0.0f;

  const int akk = tid >> 3;          // c row
  const int amm0 = (tid & 7) * 16;   // qi chunk
  const int bkk = tid >> 3;
  const int bnn0 = (tid & 7) * 16;

  for (int c0 = 0; c0 < CC; c0 += 32) {
    __syncthreads();
    {  // stage B: Bs[kk][nn] = K[c0+kk][n0+nn], contiguous -> async copy
      const _Float16* src = kb + base + (size_t)(c0 + bkk) * SS + n0 + bnn0;
      const unsigned int l0 = lds_addr(Bs + bkk * LDB_N + bnn0);
      async_copy_b128(l0, (unsigned long long)(size_t)src);
      async_copy_b128(l0 + 16, (unsigned long long)(size_t)(src + 8));
    }
    {  // stage A transposed: As[mm][kk] = Q[c0+kk][m0+mm]
      alignas(16) _Float16 tmp[16];
      const _Float16* src = qb + base + (size_t)(c0 + akk) * SS + m0 + amm0;
      *reinterpret_cast<uint4*>(tmp) = *reinterpret_cast<const uint4*>(src);
      *reinterpret_cast<uint4*>(tmp + 8) = *reinterpret_cast<const uint4*>(src + 8);
#pragma unroll
      for (int j = 0; j < 16; ++j) As[(amm0 + j) * LDA_K + akk] = tmp[j];
    }
    async_wait0();
    __syncthreads();
    wmma_tile_compute(As, Bs, acc, wm, wn, lane);
  }

  const float scale = 0.044194173824159216f;  // 1/sqrt(512)
  const int half = lane >> 4;
  const int nr = lane & 15;
  float* Sp = Sb + (size_t)zz * HW * HW;
#pragma unroll
  for (int mi = 0; mi < 2; ++mi)
#pragma unroll
    for (int ni = 0; ni < 4; ++ni)
#pragma unroll
      for (int r = 0; r < 8; ++r) {
        const int row = m0 + wm + mi * 16 + r + 8 * half;
        const int col = n0 + wn + ni * 16 + nr;
        Sp[(size_t)row * HW + col] = acc[mi][ni][r] * scale;
      }
}

// In-place row softmax over [NBT*HW] rows of length HW.
__global__ __launch_bounds__(256) void softmax_kernel(float* __restrict__ Sb) {
  float4* p = reinterpret_cast<float4*>(Sb + (size_t)blockIdx.x * HW);
  const int tid = threadIdx.x;
  float4 v = p[tid];
  __shared__ float red[256];
  float m = fmaxf(fmaxf(v.x, v.y), fmaxf(v.z, v.w));
  red[tid] = m;
  __syncthreads();
  for (int off = 128; off > 0; off >>= 1) {
    if (tid < off) red[tid] = fmaxf(red[tid], red[tid + off]);
    __syncthreads();
  }
  m = red[0];
  __syncthreads();
  v.x = __expf(v.x - m); v.y = __expf(v.y - m);
  v.z = __expf(v.z - m); v.w = __expf(v.w - m);
  red[tid] = v.x + v.y + v.z + v.w;
  __syncthreads();
  for (int off = 128; off > 0; off >>= 1) {
    if (tid < off) red[tid] += red[tid + off];
    __syncthreads();
  }
  const float inv = 1.0f / red[0];
  v.x *= inv; v.y *= inv; v.z *= inv; v.w *= inv;
  p[tid] = v;
}

// ---------------------------------------------------------------------------
// O[c][qi] = sum_k V[c][k] * P[qi][k]  (per bt batch); P fp32 -> f16 on stage.
// (A stage LDS rows are 68B-strided -> not 16B aligned; B stage needs f32->f16
//  conversion: both keep the register path.)
// ---------------------------------------------------------------------------
__global__ __launch_bounds__(256) void gemm_pv_kernel(
    const _Float16* __restrict__ vb, const float* __restrict__ Pb,
    _Float16* __restrict__ ob) {
  __shared__ __align__(16) _Float16 As[128 * LDA_K];
  __shared__ __align__(16) _Float16 Bs[32 * LDB_N];
  const int tid = threadIdx.x;
  const int lane = tid & 31;
  const int wid = tid >> 5;
  const int wm = (wid >> 1) * 32;
  const int wn = (wid & 1) * 64;
  const int n0 = blockIdx.x * 128;  // query index
  const int m0 = blockIdx.y * 128;  // channel
  const int zz = blockIdx.z;        // bt
  const size_t base = (size_t)(zz >> 4) * CC * SS + (size_t)(zz & 15) * HW;
  const float* Pz = Pb + (size_t)zz * HW * HW;

  v8f acc[2][4];
#pragma unroll
  for (int mi = 0; mi < 2; ++mi)
#pragma unroll
    for (int ni = 0; ni < 4; ++ni)
#pragma unroll
      for (int j = 0; j < 8; ++j) acc[mi][ni][j] = 0.0f;

  const int amm = tid >> 1;
  const int akk0 = (tid & 1) * 16;
  const int bnn = tid >> 3;          // 0..31 (+q*32)
  const int bkk0 = (tid & 7) * 4;

  for (int k0 = 0; k0 < HW; k0 += 32) {
    __syncthreads();
    {  // stage A: As[mm][kk] = V[m0+mm][k0+kk], contiguous K
      alignas(16) _Float16 tmp[16];
      const _Float16* src = vb + base + (size_t)(m0 + amm) * SS + k0 + akk0;
      *reinterpret_cast<uint4*>(tmp) = *reinterpret_cast<const uint4*>(src);
      *reinterpret_cast<uint4*>(tmp + 8) = *reinterpret_cast<const uint4*>(src + 8);
      _Float16* dst = As + amm * LDA_K + akk0;
#pragma unroll
      for (int j = 0; j < 16; ++j) dst[j] = tmp[j];
    }
#pragma unroll
    for (int q = 0; q < 4; ++q) {  // stage B transposed: Bs[kk][nn] = P[n0+nn][k0+kk]
      const int nn = bnn + q * 32;
      const float4 f = *reinterpret_cast<const float4*>(
          Pz + (size_t)(n0 + nn) * HW + k0 + bkk0);
      Bs[(bkk0 + 0) * LDB_N + nn] = (_Float16)f.x;
      Bs[(bkk0 + 1) * LDB_N + nn] = (_Float16)f.y;
      Bs[(bkk0 + 2) * LDB_N + nn] = (_Float16)f.z;
      Bs[(bkk0 + 3) * LDB_N + nn] = (_Float16)f.w;
    }
    __syncthreads();
    wmma_tile_compute(As, Bs, acc, wm, wn, lane);
  }

  const int half = lane >> 4;
  const int nr = lane & 15;
#pragma unroll
  for (int mi = 0; mi < 2; ++mi)
#pragma unroll
    for (int ni = 0; ni < 4; ++ni)
#pragma unroll
      for (int r = 0; r < 8; ++r) {
        const int row = m0 + wm + mi * 16 + r + 8 * half;  // channel
        const int col = n0 + wn + ni * 16 + nr;            // qi
        ob[base + (size_t)row * SS + col] = (_Float16)acc[mi][ni][r];
      }
}

// ---------------------------------------------------------------------------
extern "C" void kernel_launch(void* const* d_in, const int* in_sizes, int n_in,
                              void* d_out, int out_size, void* d_ws, size_t ws_size,
                              hipStream_t stream) {
  const float* x = (const float*)d_in[0];
  const float* gamma = (const float*)d_in[1];
  const float* beta = (const float*)d_in[2];
  const float* q_w = (const float*)d_in[3];
  const float* q_b = (const float*)d_in[4];
  const float* k_w = (const float*)d_in[5];
  const float* k_b = (const float*)d_in[6];
  const float* v_w = (const float*)d_in[7];
  const float* v_b = (const float*)d_in[8];
  const float* p_w = (const float*)d_in[9];
  const float* p_b = (const float*)d_in[10];
  float* out = (float*)d_out;

  char* ws = (char*)d_ws;
  const size_t HN_BYTES = (size_t)NBATCH * CC * SS * sizeof(_Float16);  // 32 MB
  const size_t S_BYTES = (size_t)NBT * HW * HW * sizeof(float);         // 128 MB
  float* stats = (float*)ws;                                   // 512 B
  _Float16* hn = (_Float16*)(ws + 1024);
  _Float16* qb = (_Float16*)(ws + 1024 + HN_BYTES);
  _Float16* kb = (_Float16*)(ws + 1024 + 2 * HN_BYTES);
  _Float16* vb = (_Float16*)(ws + 1024 + 3 * HN_BYTES);
  float* Sb = (float*)(ws + 1024 + 4 * HN_BYTES);
  _Float16* ab = (_Float16*)(ws + 1024 + 4 * HN_BYTES + S_BYTES);

  // 1) GroupNorm
  gn_stats_kernel<<<64, 256, 0, stream>>>(x, stats);
  gn_apply_kernel<<<(NBATCH * CC * SS) / 1024, 256, 0, stream>>>(x, gamma, beta,
                                                                 stats, hn);
  // 2) q/k/v pointwise convs (f16 out, channel-major)
  const dim3 gconv(SS / 128, CC / 128, NBATCH);
  gemm_conv_kernel<true><<<gconv, 256, 0, stream>>>(q_w, q_b, hn, qb, nullptr, nullptr);
  gemm_conv_kernel<true><<<gconv, 256, 0, stream>>>(k_w, k_b, hn, kb, nullptr, nullptr);
  gemm_conv_kernel<true><<<gconv, 256, 0, stream>>>(v_w, v_b, hn, vb, nullptr, nullptr);
  // 3) attention scores + softmax + P*V
  gemm_qk_kernel<<<dim3(HW / 128, HW / 128, NBT), 256, 0, stream>>>(qb, kb, Sb);
  softmax_kernel<<<NBT * HW, 256, 0, stream>>>(Sb);
  gemm_pv_kernel<<<dim3(HW / 128, CC / 128, NBT), 256, 0, stream>>>(vb, Sb, ab);
  // 4) projection + residual (fp32 out)
  gemm_conv_kernel<false><<<gconv, 256, 0, stream>>>(p_w, p_b, ab, nullptr, out, x);
}
